// KConv2_10960756539587
// MI455X (gfx1250) — compile-verified, matrix operand order checked by reading, not run on hardware
//
#include <hip/hip_runtime.h>

typedef _Float16 h8  __attribute__((ext_vector_type(8)));
typedef _Float16 h16 __attribute__((ext_vector_type(16)));
typedef float    f8  __attribute__((ext_vector_type(8)));

#define G_RBF 8

// ---------------------------------------------------------------------------
// Pack [Ws | Wb] -> f16, layout [Nout][Kp] (K-major per output channel),
// zero-padding K in [K, Kp).
// ---------------------------------------------------------------------------
__global__ __launch_bounds__(256) void prep_w_kernel(
    const float* __restrict__ Ws, const float* __restrict__ Wb,
    _Float16* __restrict__ BmT, int in, int K, int Kp, int Nout)
{
    int idx = blockIdx.x * blockDim.x + threadIdx.x;
    if (idx >= Nout * Kp) return;
    int o = idx / Kp;
    int k = idx % Kp;
    float v = 0.f;
    int inG = in * G_RBF;
    if (k < inG)      v = Ws[(size_t)o * inG + k];
    else if (k < K)   v = Wb[(size_t)o * in + (k - inG)];
    BmT[idx] = (_Float16)v;
}

// ---------------------------------------------------------------------------
// Patch unfold + LayerNorm + RBF basis + silu base, one wave32 per output row.
// Writes F[mrow][Kp] f16:  k = c*8+g (rbf),  in*8+c (silu),  zeros to Kp.
// ---------------------------------------------------------------------------
__global__ __launch_bounds__(256) void expand_kernel(
    const float* __restrict__ hin, const float* __restrict__ lnw,
    const float* __restrict__ lnb, _Float16* __restrict__ F,
    int Cin, int H, int W, int in, int Kp, int m0)
{
    const int lane = threadIdx.x & 31;
    const int wave = threadIdx.x >> 5;
    const int mrow = blockIdx.x * 8 + wave;     // row within chunk
    const int mg   = m0 + mrow;                 // global row = ((b*H)+y)*W+x
    const int x = mg % W;
    const int y = (mg / W) % H;
    const int b = mg / (W * H);

    const int nPer = (in + 31) >> 5;
    float pbuf[36];
    float s = 0.f, ss = 0.f;
    for (int i = 0; i < nPer; ++i) {
        int c = i * 32 + lane;
        float v = 0.f;
        if (c < in) {
            int ch = c / 9, r = c % 9;
            int yy = y + r / 3 - 1, xx = x + r % 3 - 1;
            if (yy >= 0 && yy < H && xx >= 0 && xx < W)
                v = hin[(((size_t)b * Cin + ch) * H + yy) * W + xx];
        }
        pbuf[i] = v;
        s += v; ss += v * v;
    }
    for (int off = 16; off > 0; off >>= 1) {
        s  += __shfl_xor(s,  off, 32);
        ss += __shfl_xor(ss, off, 32);
    }
    const float inv  = 1.f / (float)in;
    const float mu   = s * inv;
    const float var  = ss * inv - mu * mu;
    const float rstd = __frsqrt_rn(var + 1e-5f);

    _Float16* Frow = F + (size_t)mrow * Kp;
    const float dinv = 7.f / 4.f;               // 1/DENOM, DENOM = 4/(G-1)
    for (int i = 0; i < nPer; ++i) {
        int c = i * 32 + lane;
        if (c >= in) break;
        float p  = pbuf[i];
        float pn = (p - mu) * rstd * lnw[c] + lnb[c];
        h8 rb;
#pragma unroll
        for (int g = 0; g < G_RBF; ++g) {
            float gv = -2.f + (4.f / 7.f) * (float)g;
            float t  = (pn - gv) * dinv;
            rb[g] = (_Float16)__expf(-t * t);
        }
        *(h8*)(Frow + (size_t)c * G_RBF) = rb;               // 16B aligned
        float sil = p / (1.f + __expf(-p));
        Frow[(size_t)in * G_RBF + c] = (_Float16)sil;
    }
    for (int k = in * 9 + lane; k < Kp; k += 32)             // zero pad
        Frow[k] = (_Float16)0.f;
}

// ---------------------------------------------------------------------------
// WMMA GEMM:  Out[m][o] = sum_k F[m][k] * BmT[o][k] + bias[o]
// Block: 256 thr = 8 waves; tile 128(M) x 64(N); wave = 32x32 (2x2 wmma tiles)
// Double-buffered LDS fed by gfx1250 async global->LDS DMA (ASYNCcnt):
// one s_wait_asynccnt + one barrier per K-step, DMA overlapped with WMMA.
// ---------------------------------------------------------------------------
#define BM 128
#define BN 64
#define BK 32
#define LDA 40
#define LDB 40

__device__ inline void async_cp16(uint32_t lds_byte_addr, const void* gaddr)
{
    asm volatile("global_load_async_to_lds_b128 %0, %1, off"
                 :: "v"(lds_byte_addr),
                    "v"((unsigned long long)(uintptr_t)gaddr)
                 : "memory");
}

__device__ inline void wait_async0()
{
    asm volatile("s_wait_asynccnt 0x0" ::: "memory");
}

__device__ inline h16 ld_frag(const _Float16* base, int row, int ld, int kb)
{
    h8 lo = *(const h8*)(base + row * ld + kb);
    h8 hi = *(const h8*)(base + row * ld + 16 + kb);
    h16 r;
#pragma unroll
    for (int i = 0; i < 8; ++i) { r[i] = lo[i]; r[8 + i] = hi[i]; }
    return r;
}

__global__ __launch_bounds__(256) void gemm_kernel(
    const _Float16* __restrict__ F, const _Float16* __restrict__ BmT,
    const float* __restrict__ bias, float* __restrict__ Out,
    int Kp, int Nout)
{
    __shared__ alignas(16) _Float16 As[2][BM * LDA];
    __shared__ alignas(16) _Float16 Bs[2][BN * LDB];

    const int t    = threadIdx.x;
    const int lane = t & 31;
    const int wave = t >> 5;
    const int wm   = wave & 3;      // 4 waves along M
    const int wn   = wave >> 2;     // 2 waves along N
    const int m0   = blockIdx.x * BM;
    const int n0   = blockIdx.y * BN;

    f8 c00 = {}, c01 = {}, c10 = {}, c11 = {};

    const int ar = t >> 1;          // A: row 0..127
    const int ac = (t & 1) * 16;    //    col group (2 x 16B)
    const int br = t >> 2;          // B: n-row 0..63
    const int bc = (t & 3) * 8;     //    col (1 x 16B)

    const int kb = (lane >> 4) * 8; // fragment half-wave k base
    const int mr = lane & 15;

    // per-thread LDS destinations (byte offsets; low 32 bits of flat addr)
    const uint32_t ldsA = (uint32_t)(uintptr_t)&As[0][ar * LDA + ac];
    const uint32_t ldsB = (uint32_t)(uintptr_t)&Bs[0][br * LDB + bc];
    const uint32_t strA = (uint32_t)(sizeof(_Float16) * BM * LDA);
    const uint32_t strB = (uint32_t)(sizeof(_Float16) * BN * LDB);
    const _Float16* gA = F + (size_t)(m0 + ar) * Kp + ac;
    const _Float16* gB = BmT + (size_t)(n0 + br) * Kp + bc;

    const int nSteps = Kp / BK;

    // prologue: DMA first tiles into buffer 0
    async_cp16(ldsA,      gA);
    async_cp16(ldsA + 16, gA + 8);
    async_cp16(ldsB,      gB);

    for (int i = 0; i < nSteps; ++i) {
        const int cur = i & 1;
        wait_async0();          // my buf[cur] DMA landed
        __syncthreads();        // everyone's landed; buf[cur^1] reads retired

        if (i + 1 < nSteps) {   // prefetch next tile into the other buffer
            const int k1 = (i + 1) * BK;
            const uint32_t boff = (cur ^ 1) ? strA : 0u;
            const uint32_t boffB = (cur ^ 1) ? strB : 0u;
            async_cp16(ldsA + boff,       gA + k1);
            async_cp16(ldsA + boff + 16,  gA + k1 + 8);
            async_cp16(ldsB + boffB,      gB + k1);
        }

        const _Float16* Ab = As[cur];
        const _Float16* Bb = Bs[cur];
        h16 a0 = ld_frag(Ab, wm * 32 + mr,      LDA, kb);
        h16 a1 = ld_frag(Ab, wm * 32 + 16 + mr, LDA, kb);
        h16 b0 = ld_frag(Bb, wn * 32 + mr,      LDB, kb);
        h16 b1 = ld_frag(Bb, wn * 32 + 16 + mr, LDB, kb);

        c00 = __builtin_amdgcn_wmma_f32_16x16x32_f16(false, a0, false, b0,
                                                     (short)0, c00, false, false);
        c01 = __builtin_amdgcn_wmma_f32_16x16x32_f16(false, a0, false, b1,
                                                     (short)0, c01, false, false);
        c10 = __builtin_amdgcn_wmma_f32_16x16x32_f16(false, a1, false, b0,
                                                     (short)0, c10, false, false);
        c11 = __builtin_amdgcn_wmma_f32_16x16x32_f16(false, a1, false, b1,
                                                     (short)0, c11, false, false);
    }

    const int nc  = lane & 15;
    const int mhi = (lane >> 4) * 8;
    const float bv0 = bias[n0 + wn * 32 + nc];
    const float bv1 = bias[n0 + wn * 32 + 16 + nc];
#pragma unroll
    for (int v = 0; v < 8; ++v) {
        int m  = m0 + wm * 32 + mhi + v;
        int m2 = m + 16;
        Out[(size_t)m  * Nout + n0 + wn * 32 + nc]      = c00[v] + bv0;
        Out[(size_t)m  * Nout + n0 + wn * 32 + 16 + nc] = c01[v] + bv1;
        Out[(size_t)m2 * Nout + n0 + wn * 32 + nc]      = c10[v] + bv0;
        Out[(size_t)m2 * Nout + n0 + wn * 32 + 16 + nc] = c11[v] + bv1;
    }
}

// ---------------------------------------------------------------------------
// Per-channel sum / sumsq over NHWC Out (stats must be pre-zeroed).
// ---------------------------------------------------------------------------
__global__ __launch_bounds__(256) void bn_stats_kernel(
    const float* __restrict__ Out, float* __restrict__ stats,
    int Nrows, int Nout)
{
    int rpb = 256 / Nout;
    int o   = threadIdx.x % Nout;
    int r0  = blockIdx.x * rpb + threadIdx.x / Nout;
    int str = gridDim.x * rpb;
    float s = 0.f, ss = 0.f;
    for (int r = r0; r < Nrows; r += str) {
        float v = Out[(size_t)r * Nout + o];
        s += v; ss += v * v;
    }
    atomicAdd(&stats[o], s);
    atomicAdd(&stats[Nout + o], ss);
}

// ---------------------------------------------------------------------------
// BN (batch stats) + ReLU + 2x2 maxpool; NHWC in -> NCHW out.
// ---------------------------------------------------------------------------
__global__ __launch_bounds__(256) void bn_relu_pool_kernel(
    const float* __restrict__ Out, const float* __restrict__ stats,
    const float* __restrict__ gamma, const float* __restrict__ beta,
    float* __restrict__ hnext, int B_, int Cout, int H, int W, float invN)
{
    int Ho = H >> 1, Wo = W >> 1;
    int idx = blockIdx.x * blockDim.x + threadIdx.x;
    int total = B_ * Cout * Ho * Wo;
    if (idx >= total) return;
    int xo = idx % Wo;
    int yo = (idx / Wo) % Ho;
    int c  = (idx / (Wo * Ho)) % Cout;
    int b  = idx / (Wo * Ho * Cout);
    float mean  = stats[c] * invN;
    float var   = stats[Cout + c] * invN - mean * mean;
    float scale = gamma[c] * __frsqrt_rn(var + 1e-5f);
    float shift = beta[c] - mean * scale;
    float m = 0.f;                       // relu(v) >= 0 so 0 is a safe floor
#pragma unroll
    for (int dy = 0; dy < 2; ++dy)
#pragma unroll
        for (int dx = 0; dx < 2; ++dx) {
            int yy = yo * 2 + dy, xx = xo * 2 + dx;
            size_t row = ((size_t)b * H + yy) * W + xx;
            float v = Out[row * Cout + c] * scale + shift;
            m = fmaxf(m, fmaxf(v, 0.f));
        }
    hnext[idx] = m;                      // idx == ((b*Cout+c)*Ho+yo)*Wo+xo
}

// ---------------------------------------------------------------------------
extern "C" void kernel_launch(void* const* d_in, const int* in_sizes, int n_in,
                              void* d_out, int out_size, void* d_ws, size_t ws_size,
                              hipStream_t stream)
{
    (void)in_sizes; (void)n_in; (void)out_size; (void)ws_size;
    const int Bb = 32;

    struct LP {
        int Cin, H, W, Cout, in, K, Kp, N, Mc;
    };
    const LP lp[3] = {
        { 16, 64, 64,  64,  144,  1296,  1312, 131072, 16384 },
        { 64, 32, 32, 128,  576,  5184,  5184,  32768,  4096 },
        {128, 16, 16, 256, 1152, 10368, 10368,   8192,  2048 },
    };

    // ---- workspace carve-up (256B aligned) ----
    char* ws = (char*)d_ws;
    size_t off = 0;
    auto carve = [&](size_t bytes) -> char* {
        char* p = ws + off;
        off = (off + bytes + 255) & ~(size_t)255;
        return p;
    };
    _Float16* bmat[3];
    for (int l = 0; l < 3; ++l)
        bmat[l] = (_Float16*)carve((size_t)lp[l].Cout * lp[l].Kp * 2);
    _Float16* Fbuf  = (_Float16*)carve((size_t)16384 * 1312 * 2);   // max chunk
    float*    OutB  = (float*)carve((size_t)131072 * 64 * 4);       // max layer
    float*    h1    = (float*)carve((size_t)Bb * 64 * 32 * 32 * 4);
    float*    h2    = (float*)carve((size_t)Bb * 128 * 16 * 16 * 4);
    float*    stats = (float*)carve(2 * 256 * 4);

    const float* layer_in[3]  = { (const float*)d_in[0], h1, h2 };
    float*       layer_out[3] = { h1, h2, (float*)d_out };

    for (int l = 0; l < 3; ++l) {
        const LP& P = lp[l];
        const int base = 1 + 7 * l;
        const float* lnw = (const float*)d_in[base + 0];
        const float* lnb = (const float*)d_in[base + 1];
        const float* Wb  = (const float*)d_in[base + 2];
        const float* bb  = (const float*)d_in[base + 3];
        const float* Wsp = (const float*)d_in[base + 4];
        const float* gam = (const float*)d_in[base + 5];
        const float* bet = (const float*)d_in[base + 6];

        // 1) pack weights to f16 [Nout][Kp]
        {
            int tot = P.Cout * P.Kp;
            prep_w_kernel<<<(tot + 255) / 256, 256, 0, stream>>>(
                Wsp, Wb, bmat[l], P.in, P.K, P.Kp, P.Cout);
        }

        // 2) expand + GEMM, chunked so F stays L2-resident
        for (int m0 = 0; m0 < P.N; m0 += P.Mc) {
            expand_kernel<<<P.Mc / 8, 256, 0, stream>>>(
                layer_in[l], lnw, lnb, Fbuf, P.Cin, P.H, P.W, P.in, P.Kp, m0);
            dim3 g(P.Mc / BM, P.Cout / BN);
            gemm_kernel<<<g, 256, 0, stream>>>(
                Fbuf, bmat[l], bb, OutB + (size_t)m0 * P.Cout, P.Kp, P.Cout);
        }

        // 3) BN batch stats
        hipMemsetAsync(stats, 0, 2 * P.Cout * sizeof(float), stream);
        bn_stats_kernel<<<256, 256, 0, stream>>>(OutB, stats, P.N, P.Cout);

        // 4) BN + ReLU + maxpool -> NCHW next input (or final output)
        {
            int tot = Bb * P.Cout * (P.H / 2) * (P.W / 2);
            bn_relu_pool_kernel<<<(tot + 255) / 256, 256, 0, stream>>>(
                OutB, stats, gam, bet, layer_out[l],
                Bb, P.Cout, P.H, P.W, 1.f / (float)P.N);
        }
    }
}
